// HAttention_33079838113917
// MI455X (gfx1250) — matrix operand order repaired and checked by reading, not compiled, not run on hardware
//
#include <hip/hip_runtime.h>
#include <hip/hip_bf16.h>

// ---------------------------------------------------------------------------
// MI455X (gfx1250, wave32) implementation of per-patch multi-head attention.
// B=4, P=4, N=1024, C=512, HEADS=8, hd=64.  All GEMMs via v_wmma_f32_16x16x32_bf16.
// ---------------------------------------------------------------------------

typedef __bf16 bf16_t;
typedef __attribute__((ext_vector_type(16))) __bf16 v16bf;
typedef __attribute__((ext_vector_type(8)))  __bf16 v8bf;
typedef __attribute__((ext_vector_type(8)))  float  v8f;

#define B_  4
#define P_  4
#define N_  1024
#define C_  512
#define HEADS_ 8
#define HD_ 64

// Hardware bf16 convert (v_cvt_pk_bf16_f32) instead of software rounding.
__device__ __forceinline__ bf16_t f2bf(float f) { return (bf16_t)f; }

__device__ __forceinline__ v16bf cat16(v8bf lo, v8bf hi) {
    return __builtin_shufflevector(lo, hi, 0,1,2,3,4,5,6,7,8,9,10,11,12,13,14,15);
}

__device__ __forceinline__ v8f wmma_bf16(v16bf a, v16bf b, v8f c) {
    // (neg_a, A, neg_b, B, c_mod, C, reuse_a, reuse_b)
    return __builtin_amdgcn_wmma_f32_16x16x32_bf16(false, a, false, b, (short)0, c,
                                                   false, false);
}

// ---------------------------------------------------------------------------
// Kernel 1: fused QKV projection.  Y = X @ [Wq | Wkv],  X fp32 -> bf16 WMMA.
// grid = (3C/64, N/64, B*P), block = 128 (4 waves).  Each wave: 16 x 64 tile.
// Q is scaled by hd^-0.5 before storing (folds the softmax scale into Q).
// ---------------------------------------------------------------------------
__global__ __launch_bounds__(128)
void qkv_kernel(const float* __restrict__ x,
                const float* __restrict__ Wq,
                const float* __restrict__ Wkv,
                bf16_t* __restrict__ Qb,
                bf16_t* __restrict__ Kb,
                bf16_t* __restrict__ Vb)
{
    __shared__ bf16_t Wt[64][40];   // W chunk, transposed: [col][k], padded

    const int tid  = threadIdx.x;
    const int w    = tid >> 5;      // wave id 0..3
    const int lane = tid & 31;
    const int m    = lane & 15;     // fragment row/col within 16
    const int h    = lane >> 4;     // lane half

    const int j0 = blockIdx.x * 64;         // output column tile in [0, 3C)
    const int n0 = blockIdx.y * 64;         // row tile in [0, N)
    const int bp = blockIdx.z;              // b*P + p
    const int p  = bp & (P_ - 1);

    const float* Wb; int wstride;
    bf16_t* Db; int jloc; float oscale = 1.0f;
    if (j0 < C_) {
        Wb = Wq + (size_t)p * C_ * C_ + j0;  wstride = C_;
        Db = Qb; jloc = j0;  oscale = 0.125f;               // 1/sqrt(64)
    } else {
        Wb = Wkv + (size_t)p * C_ * 2 * C_ + (j0 - C_);  wstride = 2 * C_;
        if (j0 < 2 * C_) { Db = Kb; jloc = j0 - C_; }
        else             { Db = Vb; jloc = j0 - 2 * C_; }
    }

    const float* Arow = x + ((size_t)bp * N_ + (size_t)(n0 + w * 16 + m)) * C_;

    v8f zero = {};
    v8f acc[4] = {zero, zero, zero, zero};

    for (int kc = 0; kc < C_ / 32; ++kc) {
        const int k0 = kc * 32;

        // stage 32x64 fp32 W chunk -> bf16, transposed into LDS
        #pragma unroll
        for (int t = 0; t < 4; ++t) {
            int i   = tid + t * 128;            // 0..511 float4 chunks
            int row = i >> 4;
            int c4  = (i & 15) * 4;
            float4 v = *(const float4*)(Wb + (size_t)(k0 + row) * wstride + c4);
            Wt[c4 + 0][row] = f2bf(v.x);
            Wt[c4 + 1][row] = f2bf(v.y);
            Wt[c4 + 2][row] = f2bf(v.z);
            Wt[c4 + 3][row] = f2bf(v.w);
        }
        __syncthreads();

        // A fragment: 16x32 bf16 (lane m = row; K chunks at h*8 and h*8+16)
        const float* ap = Arow + k0 + h * 8;
        float4 f0 = *(const float4*)(ap);
        float4 f1 = *(const float4*)(ap + 4);
        float4 f2 = *(const float4*)(ap + 16);
        float4 f3 = *(const float4*)(ap + 20);
        v16bf a;
        a[0]=f2bf(f0.x); a[1]=f2bf(f0.y); a[2] =f2bf(f0.z); a[3] =f2bf(f0.w);
        a[4]=f2bf(f1.x); a[5]=f2bf(f1.y); a[6] =f2bf(f1.z); a[7] =f2bf(f1.w);
        a[8]=f2bf(f2.x); a[9]=f2bf(f2.y); a[10]=f2bf(f2.z); a[11]=f2bf(f2.w);
        a[12]=f2bf(f3.x);a[13]=f2bf(f3.y);a[14]=f2bf(f3.z); a[15]=f2bf(f3.w);

        #pragma unroll
        for (int nf = 0; nf < 4; ++nf) {
            int col = nf * 16 + m;
            v16bf b = cat16(*(const v8bf*)&Wt[col][h * 8],
                            *(const v8bf*)&Wt[col][h * 8 + 16]);
            acc[nf] = wmma_bf16(a, b, acc[nf]);
        }
        __syncthreads();
    }

    // store bf16 (C/D layout: vgpr j -> row j + h*8, lane m -> col)
    #pragma unroll
    for (int nf = 0; nf < 4; ++nf) {
        int colg = jloc + nf * 16 + m;
        #pragma unroll
        for (int j = 0; j < 8; ++j) {
            int row = n0 + w * 16 + h * 8 + j;
            Db[((size_t)bp * N_ + row) * C_ + colg] = f2bf(acc[nf][j] * oscale);
        }
    }
}

// ---------------------------------------------------------------------------
// Kernel 2: flash attention.  grid = (N/64, HEADS, B*P), block = 128 (4 waves).
// Each wave owns 16 query rows; online softmax over 16 KV tiles of 64 rows.
// Softmax reductions are split across lane-halves and combined with shfl_xor.
// ---------------------------------------------------------------------------
__global__ __launch_bounds__(128)
void attn_kernel(const bf16_t* __restrict__ Qb,
                 const bf16_t* __restrict__ Kb,
                 const bf16_t* __restrict__ Vb,
                 bf16_t* __restrict__ Ob)
{
    __shared__ bf16_t Ks[64][72];      // K tile row-major      [kv][hd]
    __shared__ bf16_t Vt[64][72];      // V tile transposed     [hd][kv]
    __shared__ float  Ss[4][16][68];   // per-wave S scratch    [row][kv]
    __shared__ bf16_t Ps[4][16][72];   // per-wave P (bf16)     [row][kv]
    __shared__ float  As[4][16];       // per-wave row rescale
    __shared__ float  Ls[4][16];       // per-wave row sum

    const int tid  = threadIdx.x;
    const int w    = tid >> 5;
    const int lane = tid & 31;
    const int m    = lane & 15;
    const int h    = lane >> 4;

    const int qt   = blockIdx.x;
    const int head = blockIdx.y;
    const int bp   = blockIdx.z;

    const size_t base = (size_t)bp * N_ * C_ + (size_t)head * HD_;
    const bf16_t* Qh = Qb + base;
    const bf16_t* Kh = Kb + base;
    const bf16_t* Vh = Vb + base;

    // Q fragments held in registers for the whole KV sweep (hd=64 -> 2 frags)
    const bf16_t* qrow = Qh + (size_t)(qt * 64 + w * 16 + m) * C_;
    v16bf qa[2];
    #pragma unroll
    for (int f = 0; f < 2; ++f)
        qa[f] = cat16(*(const v8bf*)(qrow + f * 32 + h * 16),
                      *(const v8bf*)(qrow + f * 32 + h * 16 + 8));

    v8f zero = {};
    v8f acc[4] = {zero, zero, zero, zero};
    float m_i = -3.0e38f, l_i = 0.0f;

    for (int kt = 0; kt < N_ / 64; ++kt) {
        __syncthreads();     // protect Ks/Vt from previous iteration's readers

        // stage K (row-major) and V (transposed) tiles into LDS
        #pragma unroll
        for (int t = 0; t < 4; ++t) {
            int i  = tid + t * 128;        // 0..511: 64 rows x 8 chunks of 8
            int r  = i >> 3;
            int c8 = (i & 7) * 8;
            const size_t go = (size_t)(kt * 64 + r) * C_ + c8;
            *(v8bf*)&Ks[r][c8] = *(const v8bf*)(Kh + go);
            v8bf vv = *(const v8bf*)(Vh + go);
            #pragma unroll
            for (int u = 0; u < 8; ++u) Vt[c8 + u][r] = vv[u];
        }
        if (kt + 1 < N_ / 64)   // CDNA5 prefetch of next K tile into GL2
            __builtin_prefetch(Kh + (size_t)((kt + 1) * 64 + (tid >> 1)) * C_, 0, 0);
        __syncthreads();

        // S = Q K^T  (16 x 64 per wave), two chained WMMAs over hd=64
        #pragma unroll
        for (int nf = 0; nf < 4; ++nf) {
            int n = nf * 16 + m;
            v16bf b0 = cat16(*(const v8bf*)&Ks[n][h * 16],
                             *(const v8bf*)&Ks[n][h * 16 + 8]);
            v16bf b1 = cat16(*(const v8bf*)&Ks[n][32 + h * 16],
                             *(const v8bf*)&Ks[n][32 + h * 16 + 8]);
            v8f s = zero;
            s = wmma_bf16(qa[0], b0, s);
            s = wmma_bf16(qa[1], b1, s);
            #pragma unroll
            for (int j = 0; j < 8; ++j) Ss[w][h * 8 + j][nf * 16 + m] = s[j];
        }

        // online softmax: lane handles row (lane&15), columns split by half,
        // partials combined across halves with shfl_xor(16).
        {
            const int r  = m;
            const int c0 = h * 32;
            float mt = -3.0e38f;
            #pragma unroll 8
            for (int j = 0; j < 32; ++j) mt = fmaxf(mt, Ss[w][r][c0 + j]);
            mt = fmaxf(mt, __shfl_xor(mt, 16, 32));
            float mn    = fmaxf(m_i, mt);
            float alpha = __expf(m_i - mn);
            float lsum  = 0.0f;
            #pragma unroll 8
            for (int j = 0; j < 32; ++j) {
                float pv = __expf(Ss[w][r][c0 + j] - mn);
                lsum += pv;
                Ps[w][r][c0 + j] = f2bf(pv);
            }
            lsum += __shfl_xor(lsum, 16, 32);
            l_i = l_i * alpha + lsum;
            m_i = mn;
            if (h == 0) As[w][r] = alpha;
        }

        // rescale O accumulators (rows j + h*8 in the C/D layout)
        #pragma unroll
        for (int j = 0; j < 8; ++j) {
            float aj = As[w][h * 8 + j];
            #pragma unroll
            for (int nf = 0; nf < 4; ++nf) acc[nf][j] *= aj;
        }

        // O += P V   (A = P 16x64, B = V 64x64 via transposed LDS tile)
        v16bf pa[2];
        #pragma unroll
        for (int f = 0; f < 2; ++f)
            pa[f] = cat16(*(const v8bf*)&Ps[w][m][f * 32 + h * 16],
                          *(const v8bf*)&Ps[w][m][f * 32 + h * 16 + 8]);
        #pragma unroll
        for (int nf = 0; nf < 4; ++nf) {
            int o = nf * 16 + m;
            v16bf v0 = cat16(*(const v8bf*)&Vt[o][h * 16],
                             *(const v8bf*)&Vt[o][h * 16 + 8]);
            v16bf v1 = cat16(*(const v8bf*)&Vt[o][32 + h * 16],
                             *(const v8bf*)&Vt[o][32 + h * 16 + 8]);
            acc[nf] = wmma_bf16(pa[0], v0, acc[nf]);
            acc[nf] = wmma_bf16(pa[1], v1, acc[nf]);
        }
    }

    // final 1/l normalization and bf16 store of O
    if (h == 0) Ls[w][m] = l_i;
    float inv[8];
    #pragma unroll
    for (int j = 0; j < 8; ++j) inv[j] = 1.0f / Ls[w][h * 8 + j];

    bf16_t* orow = Ob + base;
    #pragma unroll
    for (int nf = 0; nf < 4; ++nf) {
        int colg = nf * 16 + m;
        #pragma unroll
        for (int j = 0; j < 8; ++j) {
            int row = qt * 64 + w * 16 + h * 8 + j;
            orow[(size_t)row * C_ + colg] = f2bf(acc[nf][j] * inv[j]);
        }
    }
}

// ---------------------------------------------------------------------------
// Kernel 3: output projection.  out = O @ Wproj + bproj  (fp32 out).
// grid = (C/64, B*P*N/64), block = 128.
// ---------------------------------------------------------------------------
__global__ __launch_bounds__(128)
void proj_kernel(const bf16_t* __restrict__ Ob,
                 const float* __restrict__ Wp,
                 const float* __restrict__ bias,
                 float* __restrict__ out)
{
    __shared__ bf16_t Wt[64][40];

    const int tid  = threadIdx.x;
    const int w    = tid >> 5;
    const int lane = tid & 31;
    const int m    = lane & 15;
    const int h    = lane >> 4;

    const int    j0 = blockIdx.x * 64;
    const size_t r0 = (size_t)blockIdx.y * 64;
    const bf16_t* Arow = Ob + (r0 + (size_t)(w * 16 + m)) * C_;

    v8f zero = {};
    v8f acc[4] = {zero, zero, zero, zero};

    for (int kc = 0; kc < C_ / 32; ++kc) {
        const int k0 = kc * 32;
        #pragma unroll
        for (int t = 0; t < 4; ++t) {
            int i   = tid + t * 128;
            int row = i >> 4;
            int c4  = (i & 15) * 4;
            float4 v = *(const float4*)(Wp + (size_t)(k0 + row) * C_ + j0 + c4);
            Wt[c4 + 0][row] = f2bf(v.x);
            Wt[c4 + 1][row] = f2bf(v.y);
            Wt[c4 + 2][row] = f2bf(v.z);
            Wt[c4 + 3][row] = f2bf(v.w);
        }
        __syncthreads();

        v16bf a = cat16(*(const v8bf*)(Arow + k0 + h * 16),
                        *(const v8bf*)(Arow + k0 + h * 16 + 8));
        #pragma unroll
        for (int nf = 0; nf < 4; ++nf) {
            int col = nf * 16 + m;
            v16bf b = cat16(*(const v8bf*)&Wt[col][h * 16],
                            *(const v8bf*)&Wt[col][h * 16 + 8]);
            acc[nf] = wmma_bf16(a, b, acc[nf]);
        }
        __syncthreads();
    }

    #pragma unroll
    for (int nf = 0; nf < 4; ++nf) {
        int colg  = j0 + nf * 16 + m;
        float bc  = bias[colg];
        #pragma unroll
        for (int j = 0; j < 8; ++j) {
            size_t row = r0 + (size_t)(w * 16 + h * 8 + j);
            out[row * C_ + colg] = acc[nf][j] + bc;
        }
    }
}

// ---------------------------------------------------------------------------
extern "C" void kernel_launch(void* const* d_in, const int* in_sizes, int n_in,
                              void* d_out, int out_size, void* d_ws, size_t ws_size,
                              hipStream_t stream)
{
    (void)in_sizes; (void)n_in; (void)out_size; (void)ws_size;

    const float* x     = (const float*)d_in[0];
    const float* Wq    = (const float*)d_in[1];
    const float* Wkv   = (const float*)d_in[2];
    const float* Wproj = (const float*)d_in[3];
    const float* bproj = (const float*)d_in[4];

    const size_t BPNC = (size_t)B_ * P_ * N_ * C_;   // 8M elems
    bf16_t* Qb = (bf16_t*)d_ws;                      // 16 MB each
    bf16_t* Kb = Qb + BPNC;
    bf16_t* Vb = Kb + BPNC;
    bf16_t* Ob = Vb + BPNC;                          // 64 MB total scratch

    qkv_kernel <<<dim3(3 * C_ / 64, N_ / 64, B_ * P_), 128, 0, stream>>>(
        x, Wq, Wkv, Qb, Kb, Vb);
    attn_kernel<<<dim3(N_ / 64, HEADS_, B_ * P_),      128, 0, stream>>>(
        Qb, Kb, Vb, Ob);
    proj_kernel<<<dim3(C_ / 64, (B_ * P_ * N_) / 64),  128, 0, stream>>>(
        Ob, Wproj, bproj, (float*)d_out);
}
